// MultiHeadAttentionHybrid_65481071400528
// MI455X (gfx1250) — compile-verified, matrix operand order checked by reading, not run on hardware
//
#include <hip/hip_runtime.h>
#include <hip/hip_bf16.h>

#define EMBED 1024
#define HEADS 16
#define DK    64
#define BATCH 2
#define SEQ   2048
#define MTOK  (BATCH * SEQ)      // 4096 tokens
#define CHUNK_KEYS 256
#define NCHUNKS (SEQ / CHUNK_KEYS)   // 8

typedef __bf16    bf16;
typedef _Float16  f16;
typedef __attribute__((ext_vector_type(8)))  __bf16       v8bf;
typedef __attribute__((ext_vector_type(16))) __bf16       v16bf;
typedef __attribute__((ext_vector_type(8)))  _Float16     v8h;
typedef __attribute__((ext_vector_type(16))) _Float16     v16h;
typedef __attribute__((ext_vector_type(8)))  float        v8f;
typedef __attribute__((ext_vector_type(4)))  unsigned int u32x4;
typedef __attribute__((ext_vector_type(4)))  int          i32x4;
typedef __attribute__((ext_vector_type(8)))  int          i32x8;

// ---------------------------------------------------------------------------
// Fragment loader for the 16-bit WMMA A/B layout (wave32):
//   lanes 0-15  : row = lane,    K = {0..7} and {16..23}
//   lanes 16-31 : row = lane-16, K = {8..15} and {24..31}
// ---------------------------------------------------------------------------
template <typename V16, typename V8, typename T>
__device__ __forceinline__ V16 load_frag(const T* tile, int ld) {
  const int lane = threadIdx.x & 31;
  const T* p = tile + (size_t)(lane & 15) * ld + ((lane >> 4) << 3);
  V8 lo = *(const V8*)p;
  V8 hi = *(const V8*)(p + 16);
  return __builtin_shufflevector(lo, hi, 0, 1, 2, 3, 4, 5, 6, 7,
                                         8, 9, 10, 11, 12, 13, 14, 15);
}

__device__ __forceinline__ v8f wmma_bf16(v16bf a, v16bf b, v8f c) {
  return __builtin_amdgcn_wmma_f32_16x16x32_bf16(false, a, false, b,
                                                 (short)0, c, false, false);
}
__device__ __forceinline__ v8f wmma_f16(v16h a, v16h b, v8f c) {
  return __builtin_amdgcn_wmma_f32_16x16x32_f16(false, a, false, b,
                                                (short)0, c, false, false);
}

// ---------------------------------------------------------------------------
// TDM: stage a CHUNK_KEYS x 64 bf16 tile (row-major, ld=64) into LDS.
// D# per CDNA5 ISA: group0 = {count, lds_addr, global_addr,
// type=2}; group1 = {data_size=2B, tensor_dim0=64, tensor_dim1=SEQ,
// tile_dim0=64, tile_dim1=CHUNK_KEYS, dim0_stride=64, dim1_stride=64*SEQ}.
// ---------------------------------------------------------------------------
__device__ __forceinline__ void tdm_load_k_chunk(const bf16* gsrc, unsigned lds_off) {
  const unsigned long long ga = (unsigned long long)(uintptr_t)gsrc;
  u32x4 g0;
  g0.x = 1u;                                                // count=1, user D#
  g0.y = lds_off;                                           // lds_addr
  g0.z = (unsigned)(ga & 0xFFFFFFFFu);                      // global_addr[31:0]
  g0.w = (unsigned)((ga >> 32) & 0x01FFFFFFu) | (2u << 30); // addr[56:32] | type=2
  i32x8 g1 = {};
  g1[0] = 0x00010000;                          // data_size=1 (2 bytes/elem)
  g1[1] = (DK & 0xFFFF) << 16;                 // tensor_dim0 = 64 (low 16)
  g1[2] = (int)((unsigned)(SEQ & 0xFFFF) << 16);  // dim0 hi=0 | tensor_dim1 lo16
  g1[3] = DK << 16;                            // dim1 hi=0 | tile_dim0 = 64
  g1[4] = CHUNK_KEYS;                          // tile_dim1 = 256, tile_dim2 = 0
  g1[5] = DK;                                  // tensor_dim0_stride = 64 (low 32)
  g1[6] = (int)(((unsigned)(DK * SEQ) & 0xFFFFu) << 16);  // dim1_stride lo16 (=0)
  g1[7] = (DK * SEQ) >> 16;                    // dim1_stride bits[47:16]
  i32x4 gz = {};
#if __clang_major__ >= 23
  i32x8 gz8 = {};
  __builtin_amdgcn_tensor_load_to_lds(g0, g1, gz, gz, gz8, 0);
#else
  __builtin_amdgcn_tensor_load_to_lds(g0, g1, gz, gz, 0);
#endif
}

// ---------------------------------------------------------------------------
// f32 -> bf16 cast
// ---------------------------------------------------------------------------
__global__ __launch_bounds__(256) void cast_f32_bf16(const float* __restrict__ src,
                                                     bf16* __restrict__ dst, int n) {
  int i = blockIdx.x * blockDim.x + threadIdx.x;
  if (i < n) dst[i] = (bf16)src[i];
}

// ---------------------------------------------------------------------------
// C = A (M x K) * B^T (B is N x K), bf16 in, f32 acc, register-blocked:
// each wave owns a 32(M) x 64(N) tile -> 2 A-frags x 4 B-frags, 8 WMMA/K-step.
// grid = (M/32, N/256), block = 128 (4 waves split N).
// mode 0: Cb[m*N+n] bf16            mode 4: Cf[m*N+n] f32
// mode 1/2: head-split Q/K -> Cb[b][h][s][d] bf16 (Q pre-scaled)
// mode 3: head-split V transposed -> Ch[b][h][d][s] f16
// ---------------------------------------------------------------------------
__global__ __launch_bounds__(128) void gemm_bt_wmma(
    const bf16* __restrict__ A, const bf16* __restrict__ Bm,
    bf16* __restrict__ Cb, f16* __restrict__ Ch, float* __restrict__ Cf,
    int Kdim, int Ncols, int mode, float scale)
{
  const int wave = threadIdx.x >> 5;
  const int lane = threadIdx.x & 31;
  const int m0 = blockIdx.x * 32;
  const int n0 = blockIdx.y * 256 + wave * 64;

  const bf16* a0 = A  + (size_t)m0 * Kdim;
  const bf16* a1 = a0 + (size_t)16 * Kdim;
  const bf16* b0 = Bm + (size_t)n0 * Kdim;

  v8f acc[2][4];
#pragma unroll
  for (int i = 0; i < 2; ++i)
#pragma unroll
    for (int j = 0; j < 4; ++j) acc[i][j] = (v8f){};

  for (int kb = 0; kb < Kdim; kb += 32) {
    __builtin_prefetch(b0 + kb + 512, 0, 1);    // global_prefetch_b8
    v16bf af0 = load_frag<v16bf, v8bf>(a0 + kb, Kdim);
    v16bf af1 = load_frag<v16bf, v8bf>(a1 + kb, Kdim);
    v16bf bf0 = load_frag<v16bf, v8bf>(b0 + kb, Kdim);
    v16bf bf1 = load_frag<v16bf, v8bf>(b0 + (size_t)16 * Kdim + kb, Kdim);
    v16bf bf2 = load_frag<v16bf, v8bf>(b0 + (size_t)32 * Kdim + kb, Kdim);
    v16bf bf3 = load_frag<v16bf, v8bf>(b0 + (size_t)48 * Kdim + kb, Kdim);
    acc[0][0] = wmma_bf16(af0, bf0, acc[0][0]);
    acc[0][1] = wmma_bf16(af0, bf1, acc[0][1]);
    acc[0][2] = wmma_bf16(af0, bf2, acc[0][2]);
    acc[0][3] = wmma_bf16(af0, bf3, acc[0][3]);
    acc[1][0] = wmma_bf16(af1, bf0, acc[1][0]);
    acc[1][1] = wmma_bf16(af1, bf1, acc[1][1]);
    acc[1][2] = wmma_bf16(af1, bf2, acc[1][2]);
    acc[1][3] = wmma_bf16(af1, bf3, acc[1][3]);
  }

  // C/D layout: VGPR r -> (M = r + (lane>=16 ? 8 : 0), N = lane&15)
  const int nl = lane & 15;
  const int mo = (lane >> 4) << 3;
#pragma unroll
  for (int i = 0; i < 2; ++i)
#pragma unroll
    for (int j = 0; j < 4; ++j)
#pragma unroll
      for (int r = 0; r < 8; ++r) {
        const int m = m0 + i * 16 + mo + r;
        const int n = n0 + j * 16 + nl;
        const float v = acc[i][j][r] * scale;
        if (mode == 0) {
          Cb[(size_t)m * Ncols + n] = (bf16)v;
        } else if (mode == 4) {
          Cf[(size_t)m * Ncols + n] = v;
        } else {
          const int b = m >> 11, s = m & (SEQ - 1);
          const int h = n >> 6,  d = n & (DK - 1);
          if (mode == 3)
            Ch[(((size_t)b * HEADS + h) * DK + d) * SEQ + s] = (f16)v;   // V^T
          else
            Cb[(((size_t)b * HEADS + h) * SEQ + s) * DK + d] = (bf16)v;  // Q or K
        }
      }
}

// ---------------------------------------------------------------------------
// Fused attention: one block per (b, h, 16-row q-tile). block = 128 (4 waves).
//   K matrix streamed through LDS in 256-key chunks by the Tensor Data Mover
//   (double-buffered, wave 0 issues, s_wait_tensorcnt + barriers hand off).
//   GEMM1: scores(16 x 2048) = Qtile * K^T (bf16 WMMA, K from LDS) -> f16 LDS
//   softmax over 2048 keys with key-padding mask (f32 math, in-place)
//   GEMM2: out(16 x 64) = P(16x2048, LDS) * V^T(global) via f16 WMMA
// ---------------------------------------------------------------------------
__global__ __launch_bounds__(128) void attn_fused(
    const bf16* __restrict__ qh, const bf16* __restrict__ kh,
    const f16* __restrict__ vt, const int* __restrict__ mask,
    bf16* __restrict__ ao)
{
  __shared__ f16   sc[16 * SEQ];                       // 64 KB scores -> P
  __shared__ bf16  kbuf[2 * CHUNK_KEYS * DK];          // 64 KB K double-buffer
  __shared__ float red[16 * 8];
  __shared__ float rowstat[16];

  const int bh = blockIdx.x >> 7;           // b*HEADS + h  (SEQ/16 = 128 q-tiles)
  const int qt = blockIdx.x & 127;
  const int b  = bh >> 4;
  const int h  = bh & (HEADS - 1);
  const int wave = threadIdx.x >> 5;
  const int lane = threadIdx.x & 31;

  const bf16* Q    = qh + ((size_t)bh * SEQ + qt * 16) * DK;  // 16 x 64
  const bf16* Kmat = kh + (size_t)bh * SEQ * DK;              // 2048 x 64

  // Loop-invariant Q fragments (K-dim = 64 = 2 WMMA steps)
  v16bf qf0 = load_frag<v16bf, v8bf>(Q, DK);
  v16bf qf1 = load_frag<v16bf, v8bf>(Q + 32, DK);

  // ---- GEMM1 with TDM-staged K chunks ----
  if (wave == 0)
    tdm_load_k_chunk(Kmat, (unsigned)(uintptr_t)&kbuf[0]);
  for (int c = 0; c < NCHUNKS; ++c) {
    __syncthreads();   // chunk c-1 compute finished: its buffer is reusable
    if (wave == 0) {
      if (c + 1 < NCHUNKS) {
        tdm_load_k_chunk(Kmat + (size_t)(c + 1) * CHUNK_KEYS * DK,
                         (unsigned)(uintptr_t)&kbuf[((c + 1) & 1) * CHUNK_KEYS * DK]);
        __builtin_amdgcn_s_wait_tensorcnt(1);   // chunk c has landed
      } else {
        __builtin_amdgcn_s_wait_tensorcnt(0);
      }
    }
    __syncthreads();   // broadcast arrival of chunk c
    const bf16* kbase = kbuf + (size_t)(c & 1) * CHUNK_KEYS * DK;
#pragma unroll
    for (int t = 0; t < 4; ++t) {               // 4 key-tiles per wave per chunk
      const int nt = wave * 4 + t;
      const bf16* krow = kbase + (size_t)nt * 16 * DK;
      v8f acc = (v8f){};
      acc = wmma_bf16(qf0, load_frag<v16bf, v8bf>(krow, DK), acc);
      acc = wmma_bf16(qf1, load_frag<v16bf, v8bf>(krow + 32, DK), acc);
      const int n  = c * CHUNK_KEYS + nt * 16 + (lane & 15);
      const int mo = (lane >> 4) << 3;
#pragma unroll
      for (int r = 0; r < 8; ++r)
        sc[(mo + r) * SEQ + n] = (f16)acc[r];
    }
  }
  __syncthreads();

  // ---- softmax: 8 threads per row, each owns a 256-wide chunk ----
  const int row = threadIdx.x >> 3;
  const int sub = threadIdx.x & 7;
  const int CH  = SEQ / 8;
  f16* srow = sc + row * SEQ;
  const int* mrow = mask + b * SEQ;

  float mx = -3.0e38f;
  for (int j = sub * CH; j < sub * CH + CH; ++j) {
    float v = mrow[j] ? -6.0e4f : (float)srow[j];   // key-padding mask
    srow[j] = (f16)v;
    mx = fmaxf(mx, v);
  }
  red[row * 8 + sub] = mx;
  __syncthreads();
  if (sub == 0) {
    float m2 = red[row * 8];
#pragma unroll
    for (int t = 1; t < 8; ++t) m2 = fmaxf(m2, red[row * 8 + t]);
    rowstat[row] = m2;
  }
  __syncthreads();
  const float rmax = rowstat[row];
  float sum = 0.f;
  for (int j = sub * CH; j < sub * CH + CH; ++j) {
    float e = __expf((float)srow[j] - rmax);
    srow[j] = (f16)e;                   // same owner thread: safe in place
    sum += e;
  }
  red[row * 8 + sub] = sum;
  __syncthreads();
  if (sub == 0) {
    float s2 = 0.f;
#pragma unroll
    for (int t = 0; t < 8; ++t) s2 += red[row * 8 + t];
    rowstat[row] = 1.0f / s2;
  }
  __syncthreads();
  const float inv = rowstat[row];
  for (int j = sub * CH; j < sub * CH + CH; ++j)
    srow[j] = (f16)((float)srow[j] * inv);
  __syncthreads();

  // ---- GEMM2: out(16 x 64) = P * V ; wave w owns d-tile w*16 ----
  {
    const f16* vrow = vt + ((size_t)bh * DK + wave * 16) * SEQ;   // 16 x 2048
    v8f acc = (v8f){};
    for (int kb = 0; kb < SEQ; kb += 32) {
      v16h af  = load_frag<v16h, v8h>(sc + kb, SEQ);     // P from LDS
      v16h bfr = load_frag<v16h, v8h>(vrow + kb, SEQ);   // V^T from global
      acc = wmma_f16(af, bfr, acc);
    }
    const int nl = lane & 15;
    const int mo = (lane >> 4) << 3;
#pragma unroll
    for (int r = 0; r < 8; ++r) {
      const int s = qt * 16 + mo + r;
      const int e = h * DK + wave * 16 + nl;
      ao[((size_t)b * SEQ + s) * EMBED + e] = (bf16)acc[r];
    }
  }
}

// ---------------------------------------------------------------------------
// Host-side orchestration
// ---------------------------------------------------------------------------
extern "C" void kernel_launch(void* const* d_in, const int* in_sizes, int n_in,
                              void* d_out, int out_size, void* d_ws, size_t ws_size,
                              hipStream_t stream) {
  const float* x    = (const float*)d_in[0];
  const int*   mask = (const int*)d_in[1];
  const float* wq   = (const float*)d_in[2];
  const float* wk   = (const float*)d_in[3];
  const float* wv   = (const float*)d_in[4];
  const float* wo   = (const float*)d_in[5];
  float* out = (float*)d_out;

  const size_t MB = 1u << 20;
  char* ws = (char*)d_ws;
  bf16* xb  = (bf16*)(ws + 0 * MB);    // 4096x1024 bf16      (8 MB)
  bf16* wqb = (bf16*)(ws + 8 * MB);    // 1024x1024 bf16      (2 MB)
  bf16* wkb = (bf16*)(ws + 10 * MB);
  bf16* wvb = (bf16*)(ws + 12 * MB);
  bf16* wob = (bf16*)(ws + 14 * MB);
  bf16* qh  = (bf16*)(ws + 16 * MB);   // [b][h][s][d] bf16   (8 MB)
  bf16* kh  = (bf16*)(ws + 24 * MB);   // [b][h][s][d] bf16   (8 MB)
  f16*  vt  = (f16*)(ws + 32 * MB);    // [b][h][d][s] f16    (8 MB)
  bf16* ao  = (bf16*)(ws + 40 * MB);   // [b][s][e]   bf16    (8 MB)

  // 1) cast inputs to 16-bit
  {
    int n = MTOK * EMBED;
    cast_f32_bf16<<<(n + 255) / 256, 256, 0, stream>>>(x, xb, n);
    n = EMBED * EMBED;
    cast_f32_bf16<<<(n + 255) / 256, 256, 0, stream>>>(wq, wqb, n);
    cast_f32_bf16<<<(n + 255) / 256, 256, 0, stream>>>(wk, wkb, n);
    cast_f32_bf16<<<(n + 255) / 256, 256, 0, stream>>>(wv, wvb, n);
    cast_f32_bf16<<<(n + 255) / 256, 256, 0, stream>>>(wo, wob, n);
  }

  // 2) Q/K/V projections (y = x @ W^T); Q pre-scaled by 1/sqrt(DK)
  dim3 gproj(MTOK / 32, EMBED / 256);
  gemm_bt_wmma<<<gproj, 128, 0, stream>>>(xb, wqb, qh, (f16*)nullptr, (float*)nullptr,
                                          EMBED, EMBED, /*mode=*/1, 0.125f);
  gemm_bt_wmma<<<gproj, 128, 0, stream>>>(xb, wkb, kh, (f16*)nullptr, (float*)nullptr,
                                          EMBED, EMBED, /*mode=*/2, 1.0f);
  gemm_bt_wmma<<<gproj, 128, 0, stream>>>(xb, wvb, (bf16*)nullptr, vt, (float*)nullptr,
                                          EMBED, EMBED, /*mode=*/3, 1.0f);

  // 3) fused attention with TDM-staged K
  attn_fused<<<BATCH * HEADS * (SEQ / 16), 128, 0, stream>>>(qh, kh, vt, mask, ao);

  // 4) output projection -> f32 d_out
  gemm_bt_wmma<<<gproj, 128, 0, stream>>>(ao, wob, (bf16*)nullptr, (f16*)nullptr, out,
                                          EMBED, EMBED, /*mode=*/4, 1.0f);
}
